// Model_21552145891552
// MI455X (gfx1250) — matrix operand order repaired and checked by reading, not compiled, not run on hardware
//
#include <hip/hip_runtime.h>

// ---------------------------------------------------------------------------
// Transformer encoder-decoder forward for MI455X (gfx1250, wave32, WMMA).
// All GEMMs: fp32 -> bf16 operands in LDS, v_wmma_f32_16x16x32_bf16, fp32 acc.
// Loads are unguarded float4 (buffers over-allocated to tile-rounded rows,
// K always a multiple of 32); only epilogue stores are bounds-guarded.
// ---------------------------------------------------------------------------

typedef __attribute__((ext_vector_type(4)))  float          v4f;
typedef __attribute__((ext_vector_type(8)))  float          v8f;
typedef __attribute__((ext_vector_type(8)))  __bf16         v8bf;
typedef __attribute__((ext_vector_type(16))) __bf16         v16bf;
typedef __attribute__((ext_vector_type(4)))  unsigned short u16x4;
typedef __attribute__((ext_vector_type(8)))  unsigned short u16x8;

#define TB 256          // threads per block (8 waves)
#define BM 128
#define BN 128
#define BK 32
#define LDS_PITCH 40    // BK + 8 pad (bf16 units) -> 80B row stride, 16B aligned

__device__ __forceinline__ unsigned short f2bfu(float f) {
  unsigned u = __builtin_bit_cast(unsigned, f);
  u += 0x7FFFu + ((u >> 16) & 1u);           // round-to-nearest-even
  return (unsigned short)(u >> 16);
}

__device__ __forceinline__ u16x4 pack4(v4f a) {
  u16x4 r;
  r.x = f2bfu(a.x); r.y = f2bfu(a.y); r.z = f2bfu(a.z); r.w = f2bfu(a.w);
  return r;
}

__device__ __forceinline__ v16bf frag_from_lds(const unsigned short* p0,
                                               const unsigned short* p1) {
  u16x8 lo = *(const u16x8*)p0;
  u16x8 hi = *(const u16x8*)p1;
  return __builtin_shufflevector(__builtin_bit_cast(v8bf, lo),
                                 __builtin_bit_cast(v8bf, hi),
                                 0, 1, 2, 3, 4, 5, 6, 7,
                                 8, 9, 10, 11, 12, 13, 14, 15);
}

// C[M,N](ldc) = act(A[M,K](lda) @ B + bias + resid(ldc))
// TRANSB==0 : B is [K,N] row-major (ldb).  TRANSB==1 : B is [N,K] row-major (ldb).
// Batched via grid.z with element strides sA/sB/sC (resid shares sC/ldc).
// REQUIREMENTS: K % 32 == 0; lda/ldb/ldc % 4 == 0; A/B readable for the full
// tile-rounded M/N footprint (buffers over-allocated).
template <int TRANSB>
__global__ __launch_bounds__(TB) void gemm_bf16(
    const float* __restrict__ A, int lda,
    const float* __restrict__ Bm, int ldb,
    const float* __restrict__ bias, const float* __restrict__ resid,
    float* __restrict__ C, int ldc, int M, int N, int K,
    long long sA, long long sB, long long sC, int relu) {
  __shared__ __align__(16) unsigned short As[BM * LDS_PITCH];
  __shared__ __align__(16) unsigned short Bs[BN * LDS_PITCH];

  const int tid = threadIdx.x;
  const int bz  = blockIdx.z;
  A  += (long long)bz * sA;
  Bm += (long long)bz * sB;
  C  += (long long)bz * sC;
  const float* R = resid ? resid + (long long)bz * sC : nullptr;

  const int m0 = blockIdx.y * BM;
  const int n0 = blockIdx.x * BN;

  const int wave = tid >> 5;
  const int lane = tid & 31;
  const int wm   = wave & 3;    // 4 waves along M -> 32 rows each
  const int wn   = wave >> 2;   // 2 waves along N -> 64 cols each
  const int half = lane >> 4;
  const int l16  = lane & 15;

  // staging coordinates (all unguarded, 16B global accesses)
  const int ar = tid >> 3;            // 0..31  (A/transB row group)
  const int ac = (tid & 7) * 4;       // 0,4,...,28 (k sub-col)

  v8f acc[2][4];
#pragma unroll
  for (int i = 0; i < 2; ++i)
#pragma unroll
    for (int j = 0; j < 4; ++j) acc[i][j] = (v8f)(0.0f);

  for (int k0 = 0; k0 < K; k0 += BK) {
    if (k0 + BK < K)  // uniform branch; speculative prefetch of next A tile
      __builtin_prefetch(&A[(long long)(m0 + (tid & 127)) * lda + k0 + BK], 0, 1);

    // ---- stage A tile: BM x BK fp32 -> bf16, row-major [m][k] ----
#pragma unroll
    for (int it = 0; it < 4; ++it) {
      const int row = ar + it * 32;
      v4f a = *(const v4f*)&A[(long long)(m0 + row) * lda + k0 + ac];
      *(u16x4*)&As[row * LDS_PITCH + ac] = pack4(a);
    }
    // ---- stage B tile transposed into Bs[n][k] ----
    if (TRANSB) {
#pragma unroll
      for (int it = 0; it < 4; ++it) {
        const int row = ar + it * 32;   // n index
        v4f b = *(const v4f*)&Bm[(long long)(n0 + row) * ldb + k0 + ac];
        *(u16x4*)&Bs[row * LDS_PITCH + ac] = pack4(b);
      }
    } else {
#pragma unroll
      for (int it = 0; it < 4; ++it) {
        const int lin = tid + it * TB;        // 0..1023 float4 chunks
        const int kk  = lin >> 5;             // 0..31
        const int n4  = (lin & 31) * 4;       // 0..124
        v4f b = *(const v4f*)&Bm[(long long)(k0 + kk) * ldb + n0 + n4];
        Bs[(n4 + 0) * LDS_PITCH + kk] = f2bfu(b.x);
        Bs[(n4 + 1) * LDS_PITCH + kk] = f2bfu(b.y);
        Bs[(n4 + 2) * LDS_PITCH + kk] = f2bfu(b.z);
        Bs[(n4 + 3) * LDS_PITCH + kk] = f2bfu(b.w);
      }
    }
    __syncthreads();

    // ---- fragments + WMMA ----
    v16bf afrag[2], bfrag[4];
#pragma unroll
    for (int mi = 0; mi < 2; ++mi) {
      // ISA 16-bit A 16x32 layout: lane holds K runs [half*8,+8) and [16+half*8,+8)
      const unsigned short* p =
          &As[(wm * 32 + mi * 16 + l16) * LDS_PITCH + half * 8];
      afrag[mi] = frag_from_lds(p, p + 16);
    }
#pragma unroll
    for (int ni = 0; ni < 4; ++ni) {
      // ISA 16-bit B 32x16 layout: lane holds K runs [half*8,+8) and [16+half*8,+8)
      const unsigned short* p =
          &Bs[(wn * 64 + ni * 16 + l16) * LDS_PITCH + half * 8];
      bfrag[ni] = frag_from_lds(p, p + 16);
    }
#pragma unroll
    for (int mi = 0; mi < 2; ++mi)
#pragma unroll
      for (int ni = 0; ni < 4; ++ni)
        acc[mi][ni] = __builtin_amdgcn_wmma_f32_16x16x32_bf16(
            false, afrag[mi], false, bfrag[ni], (short)0, acc[mi][ni],
            false, false);
    __syncthreads();
  }

  // ---- epilogue: C/D layout: lane(n=l16, half) x VGPR r -> row r + 8*half ----
#pragma unroll
  for (int mi = 0; mi < 2; ++mi) {
#pragma unroll
    for (int ni = 0; ni < 4; ++ni) {
      const int gn = n0 + wn * 64 + ni * 16 + l16;
      const float bv = bias ? bias[gn < N ? gn : 0] : 0.0f;
#pragma unroll
      for (int r = 0; r < 8; ++r) {
        const int gm = m0 + wm * 32 + mi * 16 + r + half * 8;
        if (gm < M && gn < N) {
          float v = acc[mi][ni][r] + bv;
          if (R)    v += R[(long long)gm * ldc + gn];
          if (relu) v = fmaxf(v, 0.0f);
          C[(long long)gm * ldc + gn] = v;
        }
      }
    }
  }
}

// token embedding gather + sinusoidal positional encoding
__global__ __launch_bounds__(TB) void embed_pe_k(
    const float* __restrict__ table, const int* __restrict__ ids,
    int idsStride, float* __restrict__ out, int L, int Dm) {
  const long long row = blockIdx.x;
  const int b   = (int)(row / L);
  const int pos = (int)(row % L);
  const int id  = ids[(long long)b * idsStride + pos];
  const float* t = table + (long long)id * Dm;
  const float invD = 1.0f / (float)Dm;
  for (int d = threadIdx.x; d < Dm; d += blockDim.x) {
    const int i2 = d & ~1;
    const float div = expf(-9.210340371976184f * (float)i2 * invD); // ln(1e4)
    const float ang = (float)pos * div;
    const float pe  = (d & 1) ? cosf(ang) : sinf(ang);
    out[row * Dm + d] = t[d] + pe;
  }
}

__global__ __launch_bounds__(TB) void layernorm_k(
    const float* __restrict__ x, const float* __restrict__ g,
    const float* __restrict__ bta, float* __restrict__ y, int Dm) {
  __shared__ float s1[TB];
  __shared__ float s2[TB];
  const long long row = blockIdx.x;
  const float* xr = x + row * Dm;
  float a = 0.f, b = 0.f;
  for (int d = threadIdx.x; d < Dm; d += TB) {
    const float v = xr[d];
    a += v;
    b += v * v;
  }
  s1[threadIdx.x] = a;
  s2[threadIdx.x] = b;
  __syncthreads();
  for (int off = TB / 2; off > 0; off >>= 1) {
    if ((int)threadIdx.x < off) {
      s1[threadIdx.x] += s1[threadIdx.x + off];
      s2[threadIdx.x] += s2[threadIdx.x + off];
    }
    __syncthreads();
  }
  const float mu  = s1[0] / (float)Dm;
  const float var = s2[0] / (float)Dm - mu * mu;
  const float rs  = rsqrtf(var + 1e-5f);
  for (int d = threadIdx.x; d < Dm; d += TB)
    y[row * Dm + d] = (xr[d] - mu) * rs * g[d] + bta[d];
}

// fused scale + (optional causal mask) + softmax over one row of length Ncols,
// stored with row stride ld; zeroes the tail [limit, ld).
__global__ __launch_bounds__(TB) void softmax_k(
    float* __restrict__ att, int rowsPerB, int Ncols, int ld, long long sBt,
    float scale, int causal) {
  __shared__ float red[TB];
  const long long blk = blockIdx.x;
  const int b = (int)(blk / rowsPerB);
  const int r = (int)(blk % rowsPerB);
  float* p = att + (long long)b * sBt + (long long)r * ld;
  const int limit = causal ? (r + 1) : Ncols;

  float mx = -3.402823466e38f;
  for (int j = threadIdx.x; j < limit; j += TB) mx = fmaxf(mx, p[j] * scale);
  red[threadIdx.x] = mx;
  __syncthreads();
  for (int off = TB / 2; off > 0; off >>= 1) {
    if ((int)threadIdx.x < off)
      red[threadIdx.x] = fmaxf(red[threadIdx.x], red[threadIdx.x + off]);
    __syncthreads();
  }
  mx = red[0];
  __syncthreads();

  float sm = 0.f;
  for (int j = threadIdx.x; j < limit; j += TB) {
    const float e = expf(p[j] * scale - mx);
    p[j] = e;
    sm += e;
  }
  red[threadIdx.x] = sm;
  __syncthreads();
  for (int off = TB / 2; off > 0; off >>= 1) {
    if ((int)threadIdx.x < off) red[threadIdx.x] += red[threadIdx.x + off];
    __syncthreads();
  }
  const float inv = 1.0f / red[0];
  for (int j = threadIdx.x; j < limit; j += TB) p[j] *= inv;
  for (int j = limit + threadIdx.x; j < ld; j += TB) p[j] = 0.f;
}

// ---------------------------------------------------------------------------
// host side
// ---------------------------------------------------------------------------

static inline void launch_gemm(hipStream_t st,
                               const float* A, int lda,
                               const float* B, int ldb,
                               const float* bias, const float* resid,
                               float* C, int ldc,
                               int M, int N, int K, long long sA, long long sB,
                               long long sC, int batch, bool transB, int relu) {
  dim3 grid((N + BN - 1) / BN, (M + BM - 1) / BM, batch);
  dim3 block(TB);
  if (transB)
    gemm_bf16<1><<<grid, block, 0, st>>>(A, lda, B, ldb, bias, resid, C, ldc,
                                         M, N, K, sA, sB, sC, relu);
  else
    gemm_bf16<0><<<grid, block, 0, st>>>(A, lda, B, ldb, bias, resid, C, ldc,
                                         M, N, K, sA, sB, sC, relu);
}

// input indices (setup_inputs() dict insertion order, leaves depth-first)
enum {
  IN_SRC_TOK = 0, IN_TGT_TOK = 1,
  E_N1_G = 2,  E_N1_B = 3,
  E_WQ_W = 4,  E_WQ_B = 5,  E_WK_W = 6,  E_WK_B = 7,
  E_WV_W = 8,  E_WV_B = 9,  E_WO_W = 10, E_WO_B = 11,
  E_N2_G = 12, E_N2_B = 13,
  E_W1_W = 14, E_W1_B = 15, E_W2_W = 16, E_W2_B = 17,
  E_NO_G = 18, E_NO_B = 19,
  D_NS_G = 20, D_NS_B = 21,
  DS_WQ_W = 22, DS_WQ_B = 23, DS_WK_W = 24, DS_WK_B = 25,
  DS_WV_W = 26, DS_WV_B = 27, DS_WO_W = 28, DS_WO_B = 29,
  D_NC_G = 30, D_NC_B = 31,
  DC_WQ_W = 32, DC_WQ_B = 33, DC_WK_W = 34, DC_WK_B = 35,
  DC_WV_W = 36, DC_WV_B = 37, DC_WO_W = 38, DC_WO_B = 39,
  D_NF_G = 40, D_NF_B = 41,
  D_W1_W = 42, D_W1_B = 43, D_W2_W = 44, D_W2_B = 45,
  D_NO_G = 46, D_NO_B = 47,
  OUT_W = 48, OUT_B = 49,
  IN_SRC_NUMS = 50, IN_TGT_NUMS = 51
};

extern "C" void kernel_launch(void* const* d_in, const int* in_sizes, int n_in,
                              void* d_out, int out_size, void* d_ws, size_t ws_size,
                              hipStream_t stream) {
  constexpr int Bsz = 4, S = 1024, T = 1024, Tm1 = 1023, D = 1024, F = 4096;
  constexpr int Vt = 32000;
  const int Me = Bsz * S;     // 4096 encoder rows
  const int Md = Bsz * Tm1;   // 4092 decoder rows (tile-rounded reads hit row 4095)
  const float scale = 0.03125f;  // 1/sqrt(1024)
  const long long SD  = (long long)S * D;      // per-batch enc stride
  const long long TD  = (long long)Tm1 * D;    // per-batch dec stride
  const long long SS  = (long long)S * S;      // enc attn stride
  const long long TT  = (long long)Tm1 * 1024; // dec attn stride (ld = 1024)

  auto f32 = [&](int i) { return (const float*)d_in[i]; };
  auto i32 = [&](int i) { return (const int*)d_in[i]; };

  float* W = (float*)d_ws;
  const size_t NXE = (size_t)Me * D;   // 4,194,304 floats (= 4096 rows x 1024)
  float* bufX   = W;                   // running activations
  float* bufH   = W + NXE;             // LN output / tgt_encs
  float* bufQ   = W + 2 * NXE;         // Q, later ctx
  float* bufK   = W + 3 * NXE;         // K
  float* bufV   = W + 4 * NXE;         // V
  float* bufAtt = W + 5 * NXE;         // attention probs (ld 1024 rows)
  float* bufF   = W + 6 * NXE;         // FFN hidden (4096 rows x 4096)
  float* bufEnc = W + 6 * NXE + (size_t)Me * F;  // src_encs (persisted)
  const size_t needBytes = (7 * NXE + (size_t)Me * F) * sizeof(float);
  if (ws_size < needBytes) return;

  dim3 blk(TB);

  // ===================== encoder =====================
  embed_pe_k<<<Me, blk, 0, stream>>>(f32(IN_SRC_TOK), i32(IN_SRC_NUMS), S, bufX, S, D);

  layernorm_k<<<Me, blk, 0, stream>>>(bufX, f32(E_N1_G), f32(E_N1_B), bufH, D);

  launch_gemm(stream, bufH, D, f32(E_WQ_W), D, f32(E_WQ_B), nullptr, bufQ, D,
              Me, D, D, 0, 0, 0, 1, false, 0);
  launch_gemm(stream, bufH, D, f32(E_WK_W), D, f32(E_WK_B), nullptr, bufK, D,
              Me, D, D, 0, 0, 0, 1, false, 0);
  launch_gemm(stream, bufH, D, f32(E_WV_W), D, f32(E_WV_B), nullptr, bufV, D,
              Me, D, D, 0, 0, 0, 1, false, 0);

  // logits = Q @ K^T  (batched over B)
  launch_gemm(stream, bufQ, D, bufK, D, nullptr, nullptr, bufAtt, S,
              S, S, D, SD, SD, SS, Bsz, true, 0);
  softmax_k<<<Bsz * S, blk, 0, stream>>>(bufAtt, S, S, S, SS, scale, 0);
  // ctx = P @ V  -> reuse bufQ
  launch_gemm(stream, bufAtt, S, bufV, D, nullptr, nullptr, bufQ, D,
              S, D, S, SS, SD, SD, Bsz, false, 0);
  // x = x + ctx @ Wo + bo
  launch_gemm(stream, bufQ, D, f32(E_WO_W), D, f32(E_WO_B), bufX, bufX, D,
              Me, D, D, 0, 0, 0, 1, false, 0);

  layernorm_k<<<Me, blk, 0, stream>>>(bufX, f32(E_N2_G), f32(E_N2_B), bufH, D);
  launch_gemm(stream, bufH, D, f32(E_W1_W), F, f32(E_W1_B), nullptr, bufF, F,
              Me, F, D, 0, 0, 0, 1, false, 1);
  launch_gemm(stream, bufF, F, f32(E_W2_W), D, f32(E_W2_B), bufX, bufX, D,
              Me, D, F, 0, 0, 0, 1, false, 0);

  layernorm_k<<<Me, blk, 0, stream>>>(bufX, f32(E_NO_G), f32(E_NO_B), bufEnc, D);

  // ===================== decoder =====================
  embed_pe_k<<<Md, blk, 0, stream>>>(f32(IN_TGT_TOK), i32(IN_TGT_NUMS), T, bufX, Tm1, D);

  // --- masked self-attention ---
  layernorm_k<<<Md, blk, 0, stream>>>(bufX, f32(D_NS_G), f32(D_NS_B), bufH, D);
  launch_gemm(stream, bufH, D, f32(DS_WQ_W), D, f32(DS_WQ_B), nullptr, bufQ, D,
              Md, D, D, 0, 0, 0, 1, false, 0);
  launch_gemm(stream, bufH, D, f32(DS_WK_W), D, f32(DS_WK_B), nullptr, bufK, D,
              Md, D, D, 0, 0, 0, 1, false, 0);
  launch_gemm(stream, bufH, D, f32(DS_WV_W), D, f32(DS_WV_B), nullptr, bufV, D,
              Md, D, D, 0, 0, 0, 1, false, 0);

  // logits [1023 x 1023] stored with ld = 1024; softmax zeroes col 1023
  launch_gemm(stream, bufQ, D, bufK, D, nullptr, nullptr, bufAtt, 1024,
              Tm1, Tm1, D, TD, TD, TT, Bsz, true, 0);
  softmax_k<<<Bsz * Tm1, blk, 0, stream>>>(bufAtt, Tm1, Tm1, 1024, TT, scale, 1);
  // ctx: K = 1024 (padded prob col is exactly zero)
  launch_gemm(stream, bufAtt, 1024, bufV, D, nullptr, nullptr, bufQ, D,
              Tm1, D, 1024, TT, TD, TD, Bsz, false, 0);
  launch_gemm(stream, bufQ, D, f32(DS_WO_W), D, f32(DS_WO_B), bufX, bufX, D,
              Md, D, D, 0, 0, 0, 1, false, 0);

  // --- cross-attention (K/V from src_encs) ---
  layernorm_k<<<Md, blk, 0, stream>>>(bufX, f32(D_NC_G), f32(D_NC_B), bufH, D);
  launch_gemm(stream, bufH, D, f32(DC_WQ_W), D, f32(DC_WQ_B), nullptr, bufQ, D,
              Md, D, D, 0, 0, 0, 1, false, 0);
  launch_gemm(stream, bufEnc, D, f32(DC_WK_W), D, f32(DC_WK_B), nullptr, bufK, D,
              Me, D, D, 0, 0, 0, 1, false, 0);
  launch_gemm(stream, bufEnc, D, f32(DC_WV_W), D, f32(DC_WV_B), nullptr, bufV, D,
              Me, D, D, 0, 0, 0, 1, false, 0);

  launch_gemm(stream, bufQ, D, bufK, D, nullptr, nullptr, bufAtt, S,
              Tm1, S, D, TD, SD, (long long)Tm1 * S, Bsz, true, 0);
  softmax_k<<<Bsz * Tm1, blk, 0, stream>>>(bufAtt, Tm1, S, S,
                                           (long long)Tm1 * S, scale, 0);
  launch_gemm(stream, bufAtt, S, bufV, D, nullptr, nullptr, bufQ, D,
              Tm1, D, S, (long long)Tm1 * S, SD, TD, Bsz, false, 0);
  launch_gemm(stream, bufQ, D, f32(DC_WO_W), D, f32(DC_WO_B), bufX, bufX, D,
              Md, D, D, 0, 0, 0, 1, false, 0);

  // --- feed-forward ---
  layernorm_k<<<Md, blk, 0, stream>>>(bufX, f32(D_NF_G), f32(D_NF_B), bufH, D);
  launch_gemm(stream, bufH, D, f32(D_W1_W), F, f32(D_W1_B), nullptr, bufF, F,
              Md, F, D, 0, 0, 0, 1, false, 1);
  launch_gemm(stream, bufF, F, f32(D_W2_W), D, f32(D_W2_B), bufX, bufX, D,
              Md, D, F, 0, 0, 0, 1, false, 0);

  // --- final LN + output projection straight into d_out ---
  layernorm_k<<<Md, blk, 0, stream>>>(bufX, f32(D_NO_G), f32(D_NO_B), bufH, D);
  launch_gemm(stream, bufH, D, f32(OUT_W), Vt, f32(OUT_B), nullptr,
              (float*)d_out, Vt, Md, Vt, D, 0, 0, 0, 1, false, 0);
}
